// PhysicsInformedGNN_43379169690184
// MI455X (gfx1250) — compile-verified, matrix operand order checked by reading, not run on hardware
//
#include <hip/hip_runtime.h>
#include <hip/hip_bf16.h>

typedef __attribute__((ext_vector_type(16))) _Float16 v16h;
typedef __attribute__((ext_vector_type(8)))  _Float16 v8h;
typedef __attribute__((ext_vector_type(8)))  float    v8f;
typedef __attribute__((ext_vector_type(4)))  float    v4f;
typedef __attribute__((__vector_size__(16))) int      v4i;

#define GNN_NODES 20000
#define GNN_EDGES 320000
#define GNN_HID   256

#if __has_builtin(__builtin_amdgcn_global_load_async_to_lds_b128) && \
    __has_builtin(__builtin_amdgcn_s_wait_asynccnt)
#define GNN_ASYNC 1
#else
#define GNN_ASYNC 0
#endif

__device__ __forceinline__ void async_cp_b128(const _Float16* g, _Float16* l) {
#if GNN_ASYNC
    // generic->AS1 via integer (flat==global bits); generic->AS3 via low 32 bits
    __builtin_amdgcn_global_load_async_to_lds_b128(
        (__attribute__((address_space(1))) v4i*)(unsigned long long)g,
        (__attribute__((address_space(3))) v4i*)(unsigned int)(unsigned long long)l,
        0, 0);
#else
    *(v8h*)l = *(const v8h*)g;
#endif
}

__device__ __forceinline__ void async_wait0() {
#if GNN_ASYNC
    __builtin_amdgcn_s_wait_asynccnt(0);
#endif
}

__device__ __forceinline__ v8f zero8() {
    v8f z;
#pragma unroll
    for (int i = 0; i < 8; ++i) z[i] = 0.f;
    return z;
}

// A/B fragment for v_wmma_f32_16x16x32_f16 from a row-major f16 row.
// Caller passes p = &row[k0 + (lane>>4)*8]; fragment = p[0..7] ++ p[16..23].
__device__ __forceinline__ v16h frag_ld(const _Float16* p) {
    v8h lo = *(const v8h*)(p);
    v8h hi = *(const v8h*)(p + 16);
    v16h r;
#pragma unroll
    for (int i = 0; i < 8; ++i) { r[i] = lo[i]; r[i + 8] = hi[i]; }
    return r;
}

__device__ __forceinline__ v8f wmma_f16(v16h a, v16h b, v8f c) {
    return __builtin_amdgcn_wmma_f32_16x16x32_f16(false, a, false, b,
                                                  (short)0, c, false, false);
}

__device__ __forceinline__ void atomic_add_f32(float* p, float v) {
    __hip_atomic_fetch_add(p, v, __ATOMIC_RELAXED, __HIP_MEMORY_SCOPE_AGENT);
}

// Cooperative async fill of one 256(n) x 64(k) f16 weight slice into LDS.
// 2048 16-byte chunks over 256 threads (8 each).
__device__ __forceinline__ void fill_B64(const _Float16* __restrict__ Wt, int stride,
                                         int k0, _Float16 (*buf)[64], int tid) {
#pragma unroll
    for (int i = 0; i < 8; ++i) {
        int c = tid * 8 + i;            // 0..2047
        int n = c >> 3, seg = c & 7;    // row, 8-element segment
        async_cp_b128(Wt + (size_t)n * stride + k0 + seg * 8, &buf[n][seg * 8]);
    }
}

// Load 8 B fragments first, then run 8 WMMAs back-to-back (partial dscnt waits).
__device__ __forceinline__ void mma_row(v8f acc[8], v16h a,
                                        const _Float16 (*Bbuf)[64],
                                        int cB, int lr, int ks, int lh) {
    v16h bfr[8];
#pragma unroll
    for (int t = 0; t < 8; ++t)
        bfr[t] = frag_ld(&Bbuf[cB + t * 16 + lr][ks + lh * 8]);
#pragma unroll
    for (int t = 0; t < 8; ++t)
        acc[t] = wmma_f16(a, bfr[t], acc[t]);
}

// ---- weight prep: W[K][256] f32 -> Wt[256][Kpad] f16 (zero-padded K) ----
__global__ void k_prep_w(const float* __restrict__ W, _Float16* __restrict__ Wt,
                         int K, int Kpad) {
    int t = blockIdx.x * 256 + threadIdx.x;
    if (t >= 256 * Kpad) return;
    int n = t / Kpad, k = t % Kpad;
    Wt[t] = (k < K) ? (_Float16)W[(size_t)k * 256 + n] : (_Float16)0.f;
}

__global__ void k_zero(float* __restrict__ p, int n) {
    int i = blockIdx.x * 256 + threadIdx.x;
    if (i < n) p[i] = 0.f;
}

// ---- input projection: h = x @ Win + bin (K=6, trivial) ----
__global__ void k_input_proj(const float* __restrict__ x, const float* __restrict__ w,
                             const float* __restrict__ b, float* __restrict__ h,
                             _Float16* __restrict__ hf, int n_nodes) {
    int t = blockIdx.x * 256 + threadIdx.x;
    int node = t >> 8, j = t & 255;
    if (node >= n_nodes) return;
    float s = b[j];
#pragma unroll
    for (int c = 0; c < 6; ++c) s += x[node * 6 + c] * w[c * 256 + j];
    h[t] = s;
    hf[t] = (_Float16)s;
}

// ---- fused edge MLP: msg = L2(relu(L1([h[src]|eattr]))) ; atomic scatter to aggr[dst]
__global__ __launch_bounds__(256)
void k_edge_mlp(const _Float16* __restrict__ hf,
                const int* __restrict__ src, const int* __restrict__ dst,
                const float* __restrict__ eattr,
                const _Float16* __restrict__ Wt1, const float* __restrict__ b1,
                const _Float16* __restrict__ Wt2, const float* __restrict__ b2,
                float* __restrict__ aggr) {
    __shared__ __align__(16) _Float16 hid[64][GNN_HID];   // 32 KB
    __shared__ __align__(16) _Float16 Bsm[2][256][64];    // 2 x 32 KB, double-buffered
    const int tid = threadIdx.x;
    const int wave = tid >> 5, lane = tid & 31;
    const int rB = (wave & 3) << 4;      // 0,16,32,48
    const int cB = (wave >> 2) << 7;     // 0,128
    const int lr = lane & 15, lh = lane >> 4;
    const int e = blockIdx.x * 64 + rB + lr;              // edge row for A frags
    const _Float16* hrow = hf + (size_t)src[e] * GNN_HID;

    v8f acc[8];
#pragma unroll
    for (int t = 0; t < 8; ++t) acc[t] = zero8();

    // ---------- stage 1: Kpad = 320 (256 h + 4 eattr + zero pad) ----------
    int pp = 0;
    fill_B64(Wt1, 320, 0, Bsm[0], tid);
    async_wait0();
    __syncthreads();
#pragma unroll 1
    for (int kb = 0; kb < 256; kb += 64) {
        fill_B64(Wt1, 320, kb + 64, Bsm[pp ^ 1], tid);    // kb+64 <= 256 always valid
#pragma unroll
        for (int ks = 0; ks < 64; ks += 32) {
            v16h a = frag_ld(hrow + kb + ks + lh * 8);
            mma_row(acc, a, Bsm[pp], cB, lr, ks, lh);
        }
        async_wait0();
        __syncthreads();
        pp ^= 1;
    }
    {   // peeled tail kb=256: only k 256..287 has data (eattr + zeros); k 288..319 is all-zero -> skip
        v16h a;
        v4f ea = *(const v4f*)(eattr + (size_t)e * 4);
#pragma unroll
        for (int i = 0; i < 8; ++i) {
            a[i]     = (i < 4 && lh == 0) ? (_Float16)ea[i & 3] : (_Float16)0.f;
            a[i + 8] = (_Float16)0.f;
        }
        mma_row(acc, a, Bsm[pp], cB, lr, 0, lh);
    }
    // epilogue 1: relu(+bias) -> LDS hidden (C layout: n=lane&15, m=(lane>>4)*8+r)
#pragma unroll
    for (int t = 0; t < 8; ++t) {
        const int n = cB + t * 16 + lr;
        const float bias = b1[n];
#pragma unroll
        for (int r = 0; r < 8; ++r) {
            float v = acc[t][r] + bias;
            hid[rB + lh * 8 + r][n] = (_Float16)(v > 0.f ? v : 0.f);
        }
    }
    __syncthreads();   // all waves done reading Bsm (tail) + hid written

    // ---------- stage 2: msg = hid @ W2 + b2, scatter-add ----------
#pragma unroll
    for (int t = 0; t < 8; ++t) acc[t] = zero8();
    pp = 0;
    fill_B64(Wt2, 256, 0, Bsm[0], tid);
    async_wait0();
    __syncthreads();
#pragma unroll 1
    for (int kb = 0; kb < 256; kb += 64) {
        if (kb + 64 < 256) fill_B64(Wt2, 256, kb + 64, Bsm[pp ^ 1], tid);
#pragma unroll
        for (int ks = 0; ks < 64; ks += 32) {
            v16h a = frag_ld(&hid[rB + lr][kb + ks + lh * 8]);
            mma_row(acc, a, Bsm[pp], cB, lr, ks, lh);
        }
        async_wait0();
        __syncthreads();
        pp ^= 1;
    }
    int drow[8];
#pragma unroll
    for (int r = 0; r < 8; ++r) drow[r] = dst[blockIdx.x * 64 + rB + lh * 8 + r];
#pragma unroll
    for (int t = 0; t < 8; ++t) {
        const int n = cB + t * 16 + lr;
        const float bias = b2[n];
#pragma unroll
        for (int r = 0; r < 8; ++r)
            atomic_add_f32(aggr + (size_t)drow[r] * GNN_HID + n, acc[t][r] + bias);
    }
}

// ---- fused node-update MLP: hres = L2(relu(L1([h|aggr]))) + h ----
__global__ __launch_bounds__(256)
void k_node_mlp(const _Float16* __restrict__ hf, const float* __restrict__ aggr,
                const float* __restrict__ h32,
                const _Float16* __restrict__ Wt1, const float* __restrict__ b1,
                const _Float16* __restrict__ Wt2, const float* __restrict__ b2,
                float* __restrict__ hres, int n_nodes) {
    __shared__ __align__(16) _Float16 hid[64][GNN_HID];
    __shared__ __align__(16) _Float16 Bsm[2][256][64];
    const int tid = threadIdx.x;
    const int wave = tid >> 5, lane = tid & 31;
    const int rB = (wave & 3) << 4;
    const int cB = (wave >> 2) << 7;
    const int lr = lane & 15, lh = lane >> 4;
    const int m = blockIdx.x * 64 + rB + lr;
    const int mc = m < n_nodes ? m : n_nodes - 1;          // clamp loads only
    const _Float16* hrow = hf + (size_t)mc * GNN_HID;
    const float* arow = aggr + (size_t)mc * GNN_HID;

    v8f acc[8];
#pragma unroll
    for (int t = 0; t < 8; ++t) acc[t] = zero8();

    // ---------- stage 1: K = 512 ([h | aggr]), two branch-free halves ----------
    int pp = 0;
    fill_B64(Wt1, 512, 0, Bsm[0], tid);
    async_wait0();
    __syncthreads();
#pragma unroll 1
    for (int kb = 0; kb < 256; kb += 64) {                 // A from hf (f16)
        fill_B64(Wt1, 512, kb + 64, Bsm[pp ^ 1], tid);
#pragma unroll
        for (int ks = 0; ks < 64; ks += 32) {
            v16h a = frag_ld(hrow + kb + ks + lh * 8);
            mma_row(acc, a, Bsm[pp], cB, lr, ks, lh);
        }
        async_wait0();
        __syncthreads();
        pp ^= 1;
    }
#pragma unroll 1
    for (int kb = 256; kb < 512; kb += 64) {               // A from aggr (f32 -> f16)
        if (kb + 64 < 512) fill_B64(Wt1, 512, kb + 64, Bsm[pp ^ 1], tid);
#pragma unroll
        for (int ks = 0; ks < 64; ks += 32) {
            const float* p = arow + (kb - 256) + ks + lh * 8;
            v16h a;
#pragma unroll
            for (int i = 0; i < 8; ++i) { a[i] = (_Float16)p[i]; a[i + 8] = (_Float16)p[i + 16]; }
            mma_row(acc, a, Bsm[pp], cB, lr, ks, lh);
        }
        async_wait0();
        __syncthreads();
        pp ^= 1;
    }
#pragma unroll
    for (int t = 0; t < 8; ++t) {
        const int n = cB + t * 16 + lr;
        const float bias = b1[n];
#pragma unroll
        for (int r = 0; r < 8; ++r) {
            float v = acc[t][r] + bias;
            hid[rB + lh * 8 + r][n] = (_Float16)(v > 0.f ? v : 0.f);
        }
    }

    // ---------- stage 2 ----------
#pragma unroll
    for (int t = 0; t < 8; ++t) acc[t] = zero8();
    pp = 0;
    fill_B64(Wt2, 256, 0, Bsm[0], tid);
    async_wait0();
    __syncthreads();   // also publishes hid
#pragma unroll 1
    for (int kb = 0; kb < 256; kb += 64) {
        if (kb + 64 < 256) fill_B64(Wt2, 256, kb + 64, Bsm[pp ^ 1], tid);
#pragma unroll
        for (int ks = 0; ks < 64; ks += 32) {
            v16h a = frag_ld(&hid[rB + lr][kb + ks + lh * 8]);
            mma_row(acc, a, Bsm[pp], cB, lr, ks, lh);
        }
        async_wait0();
        __syncthreads();
        pp ^= 1;
    }
#pragma unroll
    for (int t = 0; t < 8; ++t) {
        const int n = cB + t * 16 + lr;
        const float bias = b2[n];
#pragma unroll
        for (int r = 0; r < 8; ++r) {
            int mm = blockIdx.x * 64 + rB + lh * 8 + r;
            if (mm < n_nodes)
                hres[(size_t)mm * GNN_HID + n] = acc[t][r] + bias + h32[(size_t)mm * GNN_HID + n];
        }
    }
}

// ---- layernorm over 256 channels, one wave per row; writes h f32 + f16 ----
__global__ __launch_bounds__(256)
void k_layernorm(const float* __restrict__ xin, const float* __restrict__ g,
                 const float* __restrict__ bb, float* __restrict__ h,
                 _Float16* __restrict__ hf, int n_nodes) {
    const int wave = threadIdx.x >> 5, lane = threadIdx.x & 31;
    const int row = blockIdx.x * 8 + wave;
    if (row >= n_nodes) return;
    const float* xr = xin + (size_t)row * GNN_HID;
    float v[8], s = 0.f, ss = 0.f;
#pragma unroll
    for (int i = 0; i < 8; ++i) {
        v[i] = xr[lane * 8 + i];
        s += v[i]; ss += v[i] * v[i];
    }
#pragma unroll
    for (int m = 16; m >= 1; m >>= 1) {
        s  += __shfl_xor(s, m, 32);
        ss += __shfl_xor(ss, m, 32);
    }
    const float mu  = s * (1.f / 256.f);
    const float var = ss * (1.f / 256.f) - mu * mu;
    const float inv = rsqrtf(var + 1e-5f);
#pragma unroll
    for (int i = 0; i < 8; ++i) {
        int n = lane * 8 + i;
        float y = (v[i] - mu) * inv * g[n] + bb[n];
        h[(size_t)row * GNN_HID + n]  = y;
        hf[(size_t)row * GNN_HID + n] = (_Float16)y;
    }
}

// ---- output projection: out = h @ Wout + bout (N=6) ----
__global__ void k_output_proj(const float* __restrict__ h, const float* __restrict__ w,
                              const float* __restrict__ b, float* __restrict__ out,
                              int n_nodes) {
    int t = blockIdx.x * 256 + threadIdx.x;
    if (t >= n_nodes * 6) return;
    int node = t / 6, o = t % 6;
    float s = b[o];
    const float* hr = h + (size_t)node * GNN_HID;
#pragma unroll 4
    for (int k = 0; k < GNN_HID; ++k) s += hr[k] * w[k * 6 + o];
    out[t] = s;
}

extern "C" void kernel_launch(void* const* d_in, const int* in_sizes, int n_in,
                              void* d_out, int out_size, void* d_ws, size_t ws_size,
                              hipStream_t stream) {
    (void)in_sizes; (void)n_in; (void)out_size; (void)ws_size;
    // pytree leaf order (sorted dict keys): x, edge_index, edge_attr,
    // input_proj{b,w}, layers[0..2]{ln_b,ln_g,msg.l1{b,w},msg.l2{b,w},
    // node.l1{b,w},node.l2{b,w},upd.l1{b,w},upd.l2{b,w}}, output_proj{b,w}
    const float* x     = (const float*)d_in[0];
    const int*   ei    = (const int*)d_in[1];
    const float* eattr = (const float*)d_in[2];
    const float* in_b  = (const float*)d_in[3];
    const float* in_w  = (const float*)d_in[4];
    const float* out_b = (const float*)d_in[47];
    const float* out_w = (const float*)d_in[48];
    const int* src = ei;
    const int* dst = ei + GNN_EDGES;

    char* ws = (char*)d_ws;
    size_t off = 0;
    auto take = [&](size_t bytes) {
        void* p = ws + off;
        off += (bytes + 255) & ~(size_t)255;
        return p;
    };
    float*    h    = (float*)take((size_t)GNN_NODES * GNN_HID * 4);
    _Float16* hf   = (_Float16*)take((size_t)GNN_NODES * GNN_HID * 2);
    float*    aggr = (float*)take((size_t)GNN_NODES * GNN_HID * 4);
    float*    hres = (float*)take((size_t)GNN_NODES * GNN_HID * 4);
    _Float16 *wtm1[3], *wtm2[3], *wtu1[3], *wtu2[3];
    for (int i = 0; i < 3; ++i) {
        wtm1[i] = (_Float16*)take(256ull * 320 * 2);
        wtm2[i] = (_Float16*)take(256ull * 256 * 2);
        wtu1[i] = (_Float16*)take(256ull * 512 * 2);
        wtu2[i] = (_Float16*)take(256ull * 256 * 2);
    }

    // weight prep (f32 -> transposed f16, K zero-padded)
    for (int i = 0; i < 3; ++i) {
        const int base = 5 + 14 * i;
        k_prep_w<<<(256 * 320 + 255) / 256, 256, 0, stream>>>((const float*)d_in[base + 3],  wtm1[i], 260, 320);
        k_prep_w<<<(256 * 256 + 255) / 256, 256, 0, stream>>>((const float*)d_in[base + 5],  wtm2[i], 256, 256);
        k_prep_w<<<(256 * 512 + 255) / 256, 256, 0, stream>>>((const float*)d_in[base + 11], wtu1[i], 512, 512);
        k_prep_w<<<(256 * 256 + 255) / 256, 256, 0, stream>>>((const float*)d_in[base + 13], wtu2[i], 256, 256);
    }

    k_input_proj<<<GNN_NODES, 256, 0, stream>>>(x, in_w, in_b, h, hf, GNN_NODES);

    for (int i = 0; i < 3; ++i) {
        const int base = 5 + 14 * i;
        const float* ln_b = (const float*)d_in[base + 0];
        const float* ln_g = (const float*)d_in[base + 1];
        const float* bm1  = (const float*)d_in[base + 2];
        const float* bm2  = (const float*)d_in[base + 4];
        const float* bu1  = (const float*)d_in[base + 10];
        const float* bu2  = (const float*)d_in[base + 12];
        k_zero<<<(GNN_NODES * GNN_HID + 255) / 256, 256, 0, stream>>>(aggr, GNN_NODES * GNN_HID);
        k_edge_mlp<<<GNN_EDGES / 64, 256, 0, stream>>>(hf, src, dst, eattr,
                                                       wtm1[i], bm1, wtm2[i], bm2, aggr);
        k_node_mlp<<<(GNN_NODES + 63) / 64, 256, 0, stream>>>(hf, aggr, h,
                                                              wtu1[i], bu1, wtu2[i], bu2,
                                                              hres, GNN_NODES);
        k_layernorm<<<(GNN_NODES + 7) / 8, 256, 0, stream>>>(hres, ln_g, ln_b, h, hf, GNN_NODES);
    }
    k_output_proj<<<(GNN_NODES * 6 + 255) / 256, 256, 0, stream>>>(h, out_w, out_b,
                                                                   (float*)d_out, GNN_NODES);
}
